// OnlineNcdeAligner200_61572651155896
// MI455X (gfx1250) — matrix operand order repaired and checked by reading, not compiled, not run on hardware
//
#include <hip/hip_runtime.h>
#include <math.h>

// ---------------- geometry ----------------
#define XD 200
#define YD 200
#define ZD 16
#define NXY (XD*YD)        // 40000
#define NVOX (NXY*ZD)      // 640000
#define CL 18              // logit channels
#define FC 16              // feature channels

typedef float    f4   __attribute__((ext_vector_type(4)));
typedef float    v8f  __attribute__((ext_vector_type(8)));
typedef _Float16 v16h __attribute__((ext_vector_type(16)));

union AFrag { v16h v; f4 q[2]; _Float16 h[16]; };

// ---------------- helpers ----------------
__device__ inline void loadChunk(const _Float16* __restrict__ in,
                                 int x, int y, int z, int c, int cinPad, f4* dst) {
    if ((unsigned)x < (unsigned)XD && (unsigned)y < (unsigned)YD && (unsigned)z < (unsigned)ZD) {
        const _Float16* p = in + ((size_t)((x*YD + y)*ZD + z))*cinPad + c;
        *dst = *(const f4*)p;                      // 8 x f16 = 16B
    } else {
        f4 zv = {0.f, 0.f, 0.f, 0.f};
        *dst = zv;
    }
}

// ---------------- implicit-GEMM WMMA conv (template-specialized) ----------------
// in : (X,Y,Z,CINPAD) f16    wpack : [NSTEPS][16 n][32 k] f16
// out: (X,Y,Z,16) f32 and/or f16 ; optional GN stats (8 groups x {sum,sumsq})
// ACT: 0=none 1=relu 2=tanh
template<int NSTEPS, int CINPAD, int TOTTAPS, int ACT, bool STATS>
__global__ __launch_bounds__(256)
void k_conv(const _Float16* __restrict__ in, const _Float16* __restrict__ wpack,
            const float* __restrict__ bias, int biasOff, int coutReal,
            float* __restrict__ out32, _Float16* __restrict__ out16,
            float* __restrict__ stats, int dil)
{
    __shared__ _Float16 smw[NSTEPS*512];
    const int tid = threadIdx.x;

    // ---- CDNA5 async copy: weights -> LDS (ASYNCcnt path, no VGPR bounce) ----
    constexpr int NBYTES = NSTEPS*1024;
    const unsigned ldsBase = (unsigned)(uintptr_t)(&smw[0]);   // addr[31:0] == LDS offset
    for (int i = tid*16; i < NBYTES; i += 256*16) {
        unsigned lo = ldsBase + (unsigned)i;
        unsigned long long ga = (unsigned long long)(uintptr_t)wpack + (unsigned)i;
        asm volatile("global_load_async_to_lds_b128 %0, %1, off"
                     :: "v"(lo), "v"(ga) : "memory");
    }
    asm volatile("s_wait_asynccnt 0x0" ::: "memory");
    __syncthreads();

    const int lane = tid & 31;
    const int wv   = tid >> 5;
    const int gw   = blockIdx.x*8 + wv;           // 0..39999 : one (x,y) column per wave
    const int x    = gw / YD;
    const int y    = gw % YD;
    const int row  = lane & 15;                   // M (=z) for A, N (=out ch) for B/C
    const bool hi  = lane >= 16;

    // hint the center column into cache (global_prefetch_b8)
    __builtin_prefetch(in + (size_t)(x*YD + y)*ZD*CINPAD, 0, 1);

    v8f acc = {0.f,0.f,0.f,0.f,0.f,0.f,0.f,0.f};

    #pragma unroll
    for (int s = 0; s < NSTEPS; ++s) {
        AFrag a;
        const int c0 = hi ? 8 : 0;
        if (CINPAD == 32) {
            int tap = (TOTTAPS == 1) ? 0 : s;
            int dx = 0, dy = 0, dz = 0;
            if (TOTTAPS == 27) {
                dx = (tap/9 - 1)*dil; dy = ((tap/3)%3 - 1)*dil; dz = (tap%3 - 1)*dil;
            }
            loadChunk(in, x+dx, y+dy, row+dz, c0,      32, &a.q[0]);
            loadChunk(in, x+dx, y+dy, row+dz, c0 + 16, 32, &a.q[1]);
        } else { // CINPAD == 16, two taps per K-step
            int t0 = 2*s, t1 = 2*s + 1;
            if (TOTTAPS == 1) { t0 = 0; t1 = 0; }
            else if (t1 >= TOTTAPS) t1 = t0;
            int dx0=0,dy0=0,dz0=0, dx1=0,dy1=0,dz1=0;
            if (TOTTAPS == 27) {
                dx0 = (t0/9 - 1)*dil; dy0 = ((t0/3)%3 - 1)*dil; dz0 = (t0%3 - 1)*dil;
                dx1 = (t1/9 - 1)*dil; dy1 = ((t1/3)%3 - 1)*dil; dz1 = (t1%3 - 1)*dil;
            }
            loadChunk(in, x+dx0, y+dy0, row+dz0, c0, 16, &a.q[0]);
            loadChunk(in, x+dx1, y+dy1, row+dz1, c0, 16, &a.q[1]);
        }

        AFrag b;                                   // lane holds N=row, K = 8h..8h+7, 16+8h..16+8h+7
        const _Float16* bp = &smw[s*512 + row*32 + (hi ? 8 : 0)];
        b.q[0] = *(const f4*)bp;                   // ds_load_b128
        b.q[1] = *(const f4*)(bp + 16);

        acc = __builtin_amdgcn_wmma_f32_16x16x32_f16(
                  false, a.v, false, b.v, (short)0, acc, false, false);
    }

    // epilogue: bias + activation + store (+ GN partial sums)
    const int n  = row;
    const int ng = n + biasOff;
    float bv = (bias && ng < coutReal) ? bias[ng] : 0.f;
    float ls = 0.f, lq = 0.f;
    const size_t vbase = (size_t)gw * ZD;
    #pragma unroll
    for (int r = 0; r < 8; ++r) {
        int z = r + (hi ? 8 : 0);                  // C/D layout: lanes>=16 hold M=8..15
        float val = acc[r] + bv;
        if (ACT == 1)      val = fmaxf(val, 0.f);
        else if (ACT == 2) val = tanhf(val);
        size_t o = (vbase + z)*16 + n;
        if (out32) out32[o] = val;
        if (out16) out16[o] = (_Float16)val;
        ls += val; lq += val*val;
    }
    if (STATS) {
        atomicAdd(&stats[(n>>1)*2 + 0], ls);
        atomicAdd(&stats[(n>>1)*2 + 1], lq);
    }
}

// ---------------- weight packing to B-fragment layout ----------------
__global__ void k_pack_w(const float* __restrict__ w, _Float16* __restrict__ dst,
                         int nsteps, int tapsPerStep, int totTaps,
                         int cinReal, int coutReal, int nOffset, int swapXZ)
{
    int e = blockIdx.x*256 + threadIdx.x;
    if (e >= nsteps*512) return;
    int s = e >> 9, r = e & 511, n = r >> 5, k = r & 31;
    int tap, c; bool valid = true;
    if (tapsPerStep == 1) { tap = (totTaps == 1) ? 0 : s; c = k; }
    else                  { tap = 2*s + (k >> 4);          c = k & 15; }
    if (tap >= totTaps) valid = false;
    if (c   >= cinReal) valid = false;
    int ngl = n + nOffset;
    if (ngl >= coutReal) valid = false;
    float v = 0.f;
    if (valid) {
        long idx;
        if (totTaps == 27) {
            int a = tap/9, b = (tap/3)%3, cc = tap%3;
            if (swapXZ) { int t = a; a = cc; cc = t; }   // decoder permute sandwich
            idx = ((((long)ngl*cinReal + c)*3 + a)*3 + b)*3 + cc;
        } else {
            idx = (long)ngl*cinReal + c;
        }
        v = w[idx];
    }
    dst[e] = (_Float16)v;
}

// ---------------- small / elementwise kernels ----------------
__global__ void k_zero_stats(float* s) { int e = blockIdx.x*256 + threadIdx.x; if (e < 256) s[e] = 0.f; }

__global__ void k_pack_logits(const float* __restrict__ src, _Float16* __restrict__ dst) {
    int e = blockIdx.x*256 + threadIdx.x;
    if (e >= NVOX*32) return;
    int c = e & 31, v = e >> 5;
    float val = (c < CL) ? src[(size_t)c*NVOX + v] : 0.f;
    dst[e] = (_Float16)val;
}

__global__ void k_gn_apply(const float* __restrict__ in, const float* __restrict__ stats,
                           const float* __restrict__ gamma, const float* __restrict__ beta,
                           _Float16* __restrict__ out)
{
    int e = blockIdx.x*256 + threadIdx.x;
    if (e >= NVOX*16) return;
    int n = e & 15, g = n >> 1;
    const float cnt = 2.f*(float)NVOX;
    float m  = stats[g*2+0]/cnt;
    float vr = stats[g*2+1]/cnt - m*m;
    float xn = (in[e]-m)*rsqrtf(vr + 1e-5f);
    float yv = xn*gamma[n] + beta[n];
    out[e] = (_Float16)fmaxf(yv, 0.f);            // GN is always followed by ReLU here
}

__global__ void k_sub(const float* a, const float* b, float* o) {
    int e = blockIdx.x*256 + threadIdx.x; if (e < NVOX*16) o[e] = a[e] - b[e];
}

__global__ void k_transform(const float* __restrict__ poses, const float* __restrict__ ts,
                            int k, float* __restrict__ tinv)
{
    if (threadIdx.x != 0 || blockIdx.x != 0) return;
    const float* A = poses + (size_t)k*16;        // P_k
    const float* B = poses + (size_t)(k+1)*16;    // P_{k+1}
    // Tinv = inv(transform) = inv(P_k) @ P_{k+1}, rigid: inv = [R^T | -R^T t]
    for (int i = 0; i < 3; ++i) {
        for (int j = 0; j < 3; ++j) {
            float s = 0.f;
            for (int l = 0; l < 3; ++l) s += A[l*4+i]*B[l*4+j];
            tinv[i*4+j] = s;
        }
        float s = 0.f;
        for (int l = 0; l < 3; ++l) s += A[l*4+i]*(B[l*4+3] - A[l*4+3]);
        tinv[i*4+3] = s;
    }
    tinv[12] = (ts[k+1] - ts[k]) * 1e-6f;
}

__global__ void k_warp(const float* __restrict__ Zs, const float* __restrict__ fk,
                       const float* __restrict__ fk1, const float* __restrict__ tinv,
                       float* __restrict__ Zw, float* __restrict__ fprev,
                       _Float16* __restrict__ ctrl)
{
    int v = blockIdx.x*256 + threadIdx.x;
    if (v >= NVOX) return;
    int z = v & 15, xy = v >> 4, y = xy % YD, x = xy / YD;
    float px = -40.f + (x+0.5f)*0.4f;
    float py = -40.f + (y+0.5f)*0.4f;
    float pz =  -1.f + (z+0.5f)*0.4f;
    float t0=tinv[0],t1=tinv[1],t2=tinv[2],t3=tinv[3];
    float t4=tinv[4],t5=tinv[5],t6=tinv[6],t7=tinv[7];
    float t8=tinv[8],t9=tinv[9],ta=tinv[10],tb=tinv[11];
    float qx = t0*px + t1*py + t2*pz + t3;
    float qy = t4*px + t5*py + t6*pz + t7;
    float qz = t8*px + t9*py + ta*pz + tb;
    float ix = fminf(fmaxf((qx + 40.f)*2.5f - 0.5f, 0.f), (float)(XD-1));
    float iy = fminf(fmaxf((qy + 40.f)*2.5f - 0.5f, 0.f), (float)(YD-1));
    float iz = fminf(fmaxf((qz +  1.f)*2.5f - 0.5f, 0.f), (float)(ZD-1));
    int x0 = (int)floorf(ix); float fx = ix - x0; int x1 = min(x0+1, XD-1);
    int y0 = (int)floorf(iy); float fy = iy - y0; int y1 = min(y0+1, YD-1);
    int z0 = (int)floorf(iz); float fz = iz - z0; int z1 = min(z0+1, ZD-1);
    size_t base[8]; float wgt[8];
    int cx[8] = {x0,x1,x0,x0,x1,x1,x0,x1};
    int cy[8] = {y0,y0,y1,y0,y1,y0,y1,y1};
    int cz[8] = {z0,z0,z0,z1,z0,z1,z1,z1};
    float wx[8] = {1.f-fx, fx, 1.f-fx, 1.f-fx, fx, fx, 1.f-fx, fx};
    float wy[8] = {1.f-fy, 1.f-fy, fy, 1.f-fy, fy, 1.f-fy, fy, fy};
    float wz[8] = {1.f-fz, 1.f-fz, 1.f-fz, fz, 1.f-fz, fz, fz, fz};
    for (int i = 0; i < 8; ++i) {
        base[i] = ((size_t)((cx[i]*YD + cy[i])*ZD + cz[i]))*16;
        wgt[i]  = wx[i]*wy[i]*wz[i];
    }
    float dtau = tinv[12];
    size_t vb16 = (size_t)v*16, vb32 = (size_t)v*32;
    for (int c = 0; c < 16; ++c) {
        float zw = 0.f, fp = 0.f;
        #pragma unroll
        for (int i = 0; i < 8; ++i) {
            zw += wgt[i]*Zs[base[i]+c];
            fp += wgt[i]*fk[base[i]+c];
        }
        Zw[vb16+c]    = zw;
        fprev[vb16+c] = fp;
        ctrl[vb32+c]  = (_Float16)(fk1[vb16+c] - fp);
    }
    ctrl[vb32+16] = (_Float16)dtau;
    for (int c = 17; c < 32; ++c) ctrl[vb32+c] = (_Float16)0.f;
}

__global__ void k_concat2(const float* a, const float* b, _Float16* o) {
    int e = blockIdx.x*256 + threadIdx.x;
    if (e >= NVOX*32) return;
    int c = e & 31, v = e >> 5;
    float val = (c < 16) ? a[(size_t)v*16 + c] : b[(size_t)v*16 + (c-16)];
    o[e] = (_Float16)val;
}

__global__ void k_concat_zpred(const float* Zw, const float* g1, const float* u,
                               const float* ft, _Float16* o) {
    int e = blockIdx.x*256 + threadIdx.x;
    if (e >= NVOX*32) return;
    int c = e & 31, v = e >> 5;
    float val;
    if (c < 16) { size_t i = (size_t)v*16 + c; val = Zw[i] + g1[i]*u[i]; }
    else        { val = ft[(size_t)v*16 + (c-16)]; }
    o[e] = (_Float16)val;
}

__global__ void k_heun(const float* Zw, const float* g1, const float* g2,
                       const float* u, float* Zs) {
    int e = blockIdx.x*256 + threadIdx.x;
    if (e < NVOX*16) Zs[e] = Zw[e] + 0.5f*(g1[e]+g2[e])*u[e];
}

__global__ void k_cast16(const float* a, _Float16* o) {
    int e = blockIdx.x*256 + threadIdx.x;
    if (e < NVOX*16) o[e] = (_Float16)a[e];
}

__global__ void k_final(const float* dec0, const float* dec1, const float* fast, float* out) {
    int e = blockIdx.x*256 + threadIdx.x;
    if (e >= CL*NVOX) return;
    int c = e / NVOX, v = e % NVOX;
    float val = (c < 16) ? dec0[(size_t)v*16 + c] : dec1[(size_t)v*16 + (c-16)];
    out[e] = val + fast[(size_t)2*CL*NVOX + e];
}

// ---------------- host orchestration ----------------
static inline size_t alignUp(size_t v) { return (v + 255) & ~(size_t)255; }

template<int NS, int CP, int TT, int ACT, bool ST>
static void launch_conv(const void* in, const void* wp, const float* bias, int biasOff,
                        int coutR, void* o32, void* o16, float* st, int dil,
                        hipStream_t stream) {
    k_conv<NS, CP, TT, ACT, ST><<<5000, 256, 0, stream>>>(
        (const _Float16*)in, (const _Float16*)wp, bias, biasOff, coutR,
        (float*)o32, (_Float16*)o16, st, dil);
}

extern "C" void kernel_launch(void* const* d_in, const int* in_sizes, int n_in,
                              void* d_out, int out_size, void* d_ws, size_t ws_size,
                              hipStream_t stream) {
    (void)in_sizes; (void)n_in; (void)out_size; (void)ws_size;
    const float* fast  = (const float*)d_in[0];
    const float* slow  = (const float*)d_in[1];
    const float* poses = (const float*)d_in[2];
    const float* ts    = (const float*)d_in[3];
    #define PAR(i) ((const float*)d_in[i])
    // params: 4 fe_w1,5 fe_b1,6 fe_g1,7 fe_bt1,8 fe_w2,9 fe_b2,
    //         10 se_w1,11 se_b1,12 se_g1,13 se_bt1,14 se_w2,15 se_b2,
    //         16 cp_w,17 cp_b,18 fg_win,19 fg_bin,
    //         20+4i: fg_wi, fg_bi, fg_gi, fg_bti (i=0..2),
    //         32 fg_wout,33 fg_bout,34 dec_w,35 dec_b

    char* W = (char*)d_ws;
    const size_t szF32 = (size_t)NVOX*16*4;
    const size_t szH16 = (size_t)NVOX*16*2;
    const size_t szH32 = (size_t)NVOX*32*2;
    size_t off = 0;
    auto mk = [&](size_t s){ size_t o = off; off += alignUp(s); return o; };

    size_t o_inpack = mk(szH32);
    size_t o_raw    = mk(szF32);
    size_t o_hA     = mk(szH16);
    size_t o_hB     = mk(szH16);
    size_t o_feat[3]; for (int t=0;t<3;++t) o_feat[t]=mk(szF32);
    size_t o_slow   = mk(szF32);
    size_t o_Zs = mk(szF32), o_Zw = mk(szF32), o_fp = mk(szF32);
    size_t o_u  = mk(szF32), o_g1 = mk(szF32), o_g2 = mk(szF32);
    size_t o_ctrl = mk(szH32), o_cat = mk(szH32);
    size_t o_stats = mk(1024);
    size_t o_tinv  = mk(256);
    size_t o_wfe1 = mk(27*1024), o_wse1 = mk(27*1024);
    size_t o_wfe2 = mk(1024), o_wse2 = mk(1024), o_wcp = mk(1024), o_wwin = mk(1024);
    size_t o_wb[3]; for (int i=0;i<3;++i) o_wb[i]=mk(14*1024);
    size_t o_wout = mk(1024);
    size_t o_wdec0 = mk(14*1024), o_wdec1 = mk(14*1024);
    size_t o_dec0 = o_feat[0], o_dec1 = o_feat[1];      // alias: feats dead at decode

    float* statsBase = (float*)(W + o_stats);
    float* tinv      = (float*)(W + o_tinv);

    auto packw = [&](const float* w, size_t dst, int nsteps, int tps, int tot,
                     int cinR, int coutR, int nOff, int swap) {
        int tot_e = nsteps*512;
        k_pack_w<<<(tot_e+255)/256, 256, 0, stream>>>(w, (_Float16*)(W+dst),
                 nsteps, tps, tot, cinR, coutR, nOff, swap);
    };
    const int GB16 = (NVOX*16+255)/256, GB32 = (NVOX*32+255)/256, GBV = (NVOX+255)/256;

    // ---- weight packs ----
    packw(PAR(4),  o_wfe1, 27, 1, 27, CL, 16, 0, 0);
    packw(PAR(10), o_wse1, 27, 1, 27, CL, 16, 0, 0);
    packw(PAR(8),  o_wfe2, 1, 2, 1, 16, 16, 0, 0);
    packw(PAR(14), o_wse2, 1, 2, 1, 16, 16, 0, 0);
    packw(PAR(16), o_wcp,  1, 1, 1, 17, 16, 0, 0);
    packw(PAR(18), o_wwin, 1, 1, 1, 32, 16, 0, 0);
    for (int i = 0; i < 3; ++i)
        packw(PAR(20+4*i), o_wb[i], 14, 2, 27, 16, 16, 0, 0);
    packw(PAR(32), o_wout, 1, 2, 1, 16, 16, 0, 0);
    packw(PAR(34), o_wdec0, 14, 2, 27, 16, CL, 0, 1);
    packw(PAR(34), o_wdec1, 14, 2, 27, 16, CL, 16, 1);

    k_zero_stats<<<1, 256, 0, stream>>>(statsBase);

    // ---- encoders ----
    for (int t = 0; t < 3; ++t) {
        k_pack_logits<<<GB32, 256, 0, stream>>>(fast + (size_t)t*CL*NVOX, (_Float16*)(W+o_inpack));
        launch_conv<27,32,27,0,true>(W+o_inpack, W+o_wfe1, PAR(5), 0, 16,
                                     W+o_raw, nullptr, statsBase + t*16, 1, stream);
        k_gn_apply<<<GB16, 256, 0, stream>>>((float*)(W+o_raw), statsBase + t*16,
                                             PAR(6), PAR(7), (_Float16*)(W+o_hA));
        launch_conv<1,16,1,0,false>(W+o_hA, W+o_wfe2, PAR(9), 0, 16,
                                    W+o_feat[t], nullptr, nullptr, 1, stream);
    }
    k_pack_logits<<<GB32, 256, 0, stream>>>(slow, (_Float16*)(W+o_inpack));
    launch_conv<27,32,27,0,true>(W+o_inpack, W+o_wse1, PAR(11), 0, 16,
                                 W+o_raw, nullptr, statsBase + 3*16, 1, stream);
    k_gn_apply<<<GB16, 256, 0, stream>>>((float*)(W+o_raw), statsBase + 3*16,
                                         PAR(12), PAR(13), (_Float16*)(W+o_hA));
    launch_conv<1,16,1,0,false>(W+o_hA, W+o_wse2, PAR(15), 0, 16,
                                W+o_slow, nullptr, nullptr, 1, stream);

    // Zs = slow_feat - fast_feat[0]
    k_sub<<<GB16, 256, 0, stream>>>((float*)(W+o_slow), (float*)(W+o_feat[0]), (float*)(W+o_Zs));

    // ---- Heun CDE steps ----
    auto funcg = [&](size_t gOut, int slot0) {
        launch_conv<1,32,1,1,false>(W+o_cat, W+o_wwin, PAR(19), 0, 16,
                                    nullptr, W+o_hA, nullptr, 1, stream);
        size_t cur = o_hA, nxt = o_hB;
        for (int i = 0; i < 3; ++i) {
            launch_conv<14,16,27,0,true>(W+cur, W+o_wb[i], PAR(21+4*i), 0, 16,
                                         W+o_raw, nullptr, statsBase + (slot0+i)*16,
                                         /*dil*/i+1, stream);
            k_gn_apply<<<GB16, 256, 0, stream>>>((float*)(W+o_raw), statsBase + (slot0+i)*16,
                                                 PAR(22+4*i), PAR(23+4*i), (_Float16*)(W+nxt));
            size_t tmp = cur; cur = nxt; nxt = tmp;
        }
        launch_conv<1,16,1,2,false>(W+cur, W+o_wout, PAR(33), 0, 16,
                                    W+gOut, nullptr, nullptr, 1, stream);
    };

    for (int k = 0; k < 2; ++k) {
        k_transform<<<1, 1, 0, stream>>>(poses, ts, k, tinv);
        k_warp<<<GBV, 256, 0, stream>>>((float*)(W+o_Zs), (float*)(W+o_feat[k]),
                                        (float*)(W+o_feat[k+1]), tinv,
                                        (float*)(W+o_Zw), (float*)(W+o_fp),
                                        (_Float16*)(W+o_ctrl));
        launch_conv<1,32,1,0,false>(W+o_ctrl, W+o_wcp, PAR(17), 0, 16,
                                    W+o_u, nullptr, nullptr, 1, stream);
        k_concat2<<<GB32, 256, 0, stream>>>((float*)(W+o_Zw), (float*)(W+o_fp), (_Float16*)(W+o_cat));
        funcg(o_g1, 4 + (k*2+0)*3);
        k_concat_zpred<<<GB32, 256, 0, stream>>>((float*)(W+o_Zw), (float*)(W+o_g1),
                                                 (float*)(W+o_u), (float*)(W+o_feat[k+1]),
                                                 (_Float16*)(W+o_cat));
        funcg(o_g2, 4 + (k*2+1)*3);
        k_heun<<<GB16, 256, 0, stream>>>((float*)(W+o_Zw), (float*)(W+o_g1),
                                         (float*)(W+o_g2), (float*)(W+o_u), (float*)(W+o_Zs));
    }

    // ---- decoder (axes-swapped 3x3x3 conv, 18 out channels = two N tiles) ----
    k_cast16<<<GB16, 256, 0, stream>>>((float*)(W+o_Zs), (_Float16*)(W+o_hA));
    launch_conv<14,16,27,0,false>(W+o_hA, W+o_wdec0, PAR(35), 0,  CL,
                                  W+o_dec0, nullptr, nullptr, 1, stream);
    launch_conv<14,16,27,0,false>(W+o_hA, W+o_wdec1, PAR(35), 16, CL,
                                  W+o_dec1, nullptr, nullptr, 1, stream);
    k_final<<<(CL*NVOX+255)/256, 256, 0, stream>>>((float*)(W+o_dec0), (float*)(W+o_dec1),
                                                   fast, (float*)d_out);
}
